// Llama4TextMoe_53798760349864
// MI455X (gfx1250) — matrix operand order repaired and checked by reading, not compiled, not run on hardware
//
#include <hip/hip_runtime.h>
#include <hip/hip_bf16.h>
#include <stdint.h>

// Problem constants (from reference): E=8, H=1024, I=4096, T=2*1024=2048, TOPK=1
#define T_TOK 2048
#define H_DIM 1024
#define I_DIM 4096
#define E_NUM 8

typedef __attribute__((ext_vector_type(16))) __bf16 v16bf;
typedef __attribute__((ext_vector_type(8)))  __bf16 v8bf;
typedef __attribute__((ext_vector_type(8)))  float  v8f;
typedef __attribute__((ext_vector_type(4)))  int    v4i;

#if __has_builtin(__builtin_amdgcn_global_load_async_to_lds_b128)
#define ASYNC_LDS_COPY 1
#else
#define ASYNC_LDS_COPY 0
#endif

__device__ __forceinline__ v16bf cat8(v8bf lo, v8bf hi) {
  return __builtin_shufflevector(lo, hi, 0,1,2,3,4,5,6,7,8,9,10,11,12,13,14,15);
}

// Pack two fp32 -> packed bf16 pair (low = a, high = b).
__device__ __forceinline__ uint32_t pack_bf16x2(float a, float b) {
  union { __bf16 h[2]; uint32_t u; } v;
  v.h[0] = (__bf16)a;
  v.h[1] = (__bf16)b;
  return v.u;
}

// 16B global -> LDS copy; async (ASYNCcnt-tracked, bypasses VGPRs) when available.
__device__ __forceinline__ void copy_b128_to_lds(const __bf16* gsrc, __bf16* lds_dst) {
#if ASYNC_LDS_COPY
  __builtin_amdgcn_global_load_async_to_lds_b128(
      (__attribute__((address_space(1))) v4i*)gsrc,
      (__attribute__((address_space(3))) v4i*)lds_dst, 0, 0);
#else
  *(v8bf*)lds_dst = *(const v8bf*)gsrc;
#endif
}

__device__ __forceinline__ void wait_async_lds() {
#if ASYNC_LDS_COPY
#if __has_builtin(__builtin_amdgcn_s_wait_asynccnt)
  __builtin_amdgcn_s_wait_asynccnt(0);
#else
  asm volatile("s_wait_asynccnt 0" ::: "memory");
#endif
#endif
}

// ---------------------------------------------------------------------------
// Kernel 1: router. One wave (32 lanes) per token: logits = x @ router_w (H x 8),
// top-1, sigmoid. Writes scores[T] (ws) and router_scores.T (E x T) into d_out tail.
// ---------------------------------------------------------------------------
__global__ __launch_bounds__(256) void router_kernel(
    const float* __restrict__ x, const float* __restrict__ rw,
    float* __restrict__ scores, float* __restrict__ scoresT) {
  const int lane = threadIdx.x & 31;
  const int wave = threadIdx.x >> 5;
  const int t = blockIdx.x * 8 + wave;
  if (t >= T_TOK) return;
  const float* xr = x + (size_t)t * H_DIM;
  float acc[E_NUM];
#pragma unroll
  for (int e = 0; e < E_NUM; ++e) acc[e] = 0.f;
  for (int h = lane; h < H_DIM; h += 32) {
    const float xv = xr[h];
    const float* wr = rw + (size_t)h * E_NUM;
#pragma unroll
    for (int e = 0; e < E_NUM; ++e) acc[e] += xv * wr[e];
  }
#pragma unroll
  for (int off = 16; off > 0; off >>= 1) {
#pragma unroll
    for (int e = 0; e < E_NUM; ++e) acc[e] += __shfl_xor(acc[e], off, 32);
  }
  if (lane == 0) {
    int imax = 0;
    float mx = acc[0];
#pragma unroll
    for (int e = 1; e < E_NUM; ++e) {
      if (acc[e] > mx) { mx = acc[e]; imax = e; }
    }
    const float sig = __builtin_amdgcn_rcpf(1.f + __expf(-mx));
    scores[t] = sig;
#pragma unroll
    for (int e = 0; e < E_NUM; ++e)
      scoresT[(size_t)e * T_TOK + t] = (e == imax) ? sig : 0.f;
  }
}

// ---------------------------------------------------------------------------
// Kernel 2: convert activations to bf16: x16 = bf16(x), xs16 = bf16(x * score[t]).
// ---------------------------------------------------------------------------
__global__ __launch_bounds__(256) void convert_kernel(
    const float* __restrict__ x, const float* __restrict__ scores,
    __bf16* __restrict__ x16, __bf16* __restrict__ xs16) {
  const size_t i = ((size_t)blockIdx.x * blockDim.x + threadIdx.x) * 4;
  const int t = (int)(i >> 10);  // H_DIM == 1024
  const float s = scores[t];
  const float4 v = *(const float4*)(x + i);
  x16[i + 0] = (__bf16)v.x;
  x16[i + 1] = (__bf16)v.y;
  x16[i + 2] = (__bf16)v.z;
  x16[i + 3] = (__bf16)v.w;
  xs16[i + 0] = (__bf16)(v.x * s);
  xs16[i + 1] = (__bf16)(v.y * s);
  xs16[i + 2] = (__bf16)(v.z * s);
  xs16[i + 3] = (__bf16)(v.w * s);
}

// ---------------------------------------------------------------------------
// GEMM1: fused gate+up projection + SwiGLU.
//   A:  (T, H) bf16 row-major (x16 for shared, xs16 for experts)
//   Bg/Bu: fp32 (K=H, N) row-major with row stride ldb, per-chunk offset estride
//   act: (T, I) bf16 = silu(A@Bg) * (A@Bu)
// Block: 256 thr (8 waves), tile M=128, N=64, K-step 32. Waves: 4(M) x 2(N),
// each wave owns a 32x32 tile of BOTH gate and up outputs (8 v8f accumulators).
// A tile staged global->LDS via async-to-LDS; fp32 weights converted to packed
// bf16 pairs while staging into transposed, padded LDS.
// ---------------------------------------------------------------------------
#define LDA 40  // 32 + 8 pad (keeps 16B alignment of b128 fragments)

__global__ __launch_bounds__(256) void gemm_gateup_kernel(
    const __bf16* __restrict__ A,
    const float* __restrict__ Bg_base,
    const float* __restrict__ Bu_base,
    int64_t ldb, int64_t estride,
    __bf16* __restrict__ act) {
  const int tid  = threadIdx.x;
  const int lane = tid & 31;
  const int wave = tid >> 5;
  const int wm = wave >> 1;   // 0..3
  const int wn = wave & 1;    // 0..1
  const int m0 = blockIdx.x * 128;
  const int n0 = blockIdx.y * 64;
  const int chunk = m0 >> 8;  // 256-token expert chunk
  const float* Bg = Bg_base + (int64_t)chunk * estride;
  const float* Bu = Bu_base + (int64_t)chunk * estride;

  __shared__ __align__(16) __bf16 As [128 * LDA];  // [m][k]
  __shared__ __align__(16) __bf16 Bgs[ 64 * LDA];  // [n][k] (transposed)
  __shared__ __align__(16) __bf16 Bus[ 64 * LDA];

  v8f accg[2][2] = {};
  v8f accu[2][2] = {};

  const int l16 = lane & 15;
  const int sel = lane >> 4;

  for (int k0 = 0; k0 < H_DIM; k0 += 32) {
    __syncthreads();
    // ---- stage A tile (128x32 bf16): async global->LDS, 16B per lane
#pragma unroll
    for (int it = 0; it < 2; ++it) {
      const int row = (tid >> 2) + it * 64;
      const int c8  = (tid & 3) * 8;
      copy_b128_to_lds(A + (int64_t)(m0 + row) * H_DIM + k0 + c8,
                       &As[row * LDA + c8]);
    }
    // ---- stage B tiles (32x64 fp32 -> bf16 pairs, transposed into [n][k])
    {
      const int k2 = (tid >> 4) * 2;   // 0,2,...,30
      const int n  = (tid & 15) * 4;   // 0,4,...,60
      const int64_t g0 = (int64_t)(k0 + k2) * ldb + n0 + n;
      const float4 ga = *(const float4*)(Bg + g0);
      const float4 gb = *(const float4*)(Bg + g0 + ldb);
      const float4 ua = *(const float4*)(Bu + g0);
      const float4 ub = *(const float4*)(Bu + g0 + ldb);
      *(uint32_t*)(&Bgs[(n + 0) * LDA + k2]) = pack_bf16x2(ga.x, gb.x);
      *(uint32_t*)(&Bgs[(n + 1) * LDA + k2]) = pack_bf16x2(ga.y, gb.y);
      *(uint32_t*)(&Bgs[(n + 2) * LDA + k2]) = pack_bf16x2(ga.z, gb.z);
      *(uint32_t*)(&Bgs[(n + 3) * LDA + k2]) = pack_bf16x2(ga.w, gb.w);
      *(uint32_t*)(&Bus[(n + 0) * LDA + k2]) = pack_bf16x2(ua.x, ub.x);
      *(uint32_t*)(&Bus[(n + 1) * LDA + k2]) = pack_bf16x2(ua.y, ub.y);
      *(uint32_t*)(&Bus[(n + 2) * LDA + k2]) = pack_bf16x2(ua.z, ub.z);
      *(uint32_t*)(&Bus[(n + 3) * LDA + k2]) = pack_bf16x2(ua.w, ub.w);
      if (k0 + 32 < H_DIM) {  // hide HBM latency on streaming fp32 weights
        __builtin_prefetch(Bg + g0 + 32 * ldb, 0, 1);
        __builtin_prefetch(Bu + g0 + 32 * ldb, 0, 1);
      }
    }
    wait_async_lds();
    __syncthreads();

    // ---- fragments (CDNA5 16-bit WMMA layouts, ISA 7.12.2)
    // A 16x32: lane<16 holds K 0..7,16..23 ; lane>=16 holds K 8..15,24..31
    v16bf afrag[2];
#pragma unroll
    for (int mi = 0; mi < 2; ++mi) {
      const int base = (wm * 32 + mi * 16 + l16) * LDA + sel * 8;
      afrag[mi] = cat8(*(const v8bf*)(&As[base]), *(const v8bf*)(&As[base + 16]));
    }
    // B 32x16: lanes 0-15 -> K 0..7,16..23 ; lanes 16-31 -> K 8..15,24..31, N = lane%16
    v16bf bgf[2], buf2[2];
#pragma unroll
    for (int ni = 0; ni < 2; ++ni) {
      const int base = (wn * 32 + ni * 16 + l16) * LDA + sel * 8;
      bgf[ni]  = cat8(*(const v8bf*)(&Bgs[base]), *(const v8bf*)(&Bgs[base + 16]));
      buf2[ni] = cat8(*(const v8bf*)(&Bus[base]), *(const v8bf*)(&Bus[base + 16]));
    }
#pragma unroll
    for (int mi = 0; mi < 2; ++mi) {
#pragma unroll
      for (int ni = 0; ni < 2; ++ni) {
        accg[mi][ni] = __builtin_amdgcn_wmma_f32_16x16x32_bf16(
            false, afrag[mi], false, bgf[ni], (short)0, accg[mi][ni], false, false);
        accu[mi][ni] = __builtin_amdgcn_wmma_f32_16x16x32_bf16(
            false, afrag[mi], false, buf2[ni], (short)0, accu[mi][ni], false, false);
      }
    }
  }

  // ---- fused SwiGLU epilogue: act = silu(gate) * up  -> bf16 (fast v_rcp_f32)
#pragma unroll
  for (int mi = 0; mi < 2; ++mi) {
#pragma unroll
    for (int ni = 0; ni < 2; ++ni) {
#pragma unroll
      for (int r = 0; r < 8; ++r) {
        const float gv = accg[mi][ni][r];
        const float uv = accu[mi][ni][r];
        const float a = gv * __builtin_amdgcn_rcpf(1.f + __expf(-gv)) * uv;
        const int m = m0 + wm * 32 + mi * 16 + r + 8 * sel;
        const int n = n0 + wn * 32 + ni * 16 + l16;
        act[(int64_t)m * I_DIM + n] = (__bf16)a;
      }
    }
  }
}

// ---------------------------------------------------------------------------
// GEMM2: down projection, two K-phases (shared then expert) summed into one
// fp32 output tile. A: (T, I) bf16, B: (I, H) fp32 (expert chunk-offset).
// ---------------------------------------------------------------------------
__global__ __launch_bounds__(256) void gemm_down_kernel(
    const __bf16* __restrict__ act_sh, const __bf16* __restrict__ act_ex,
    const float* __restrict__ Wsh, const float* __restrict__ Wexp_base,
    float* __restrict__ out) {
  const int tid  = threadIdx.x;
  const int lane = tid & 31;
  const int wave = tid >> 5;
  const int wm = wave >> 1;
  const int wn = wave & 1;
  const int m0 = blockIdx.x * 128;
  const int n0 = blockIdx.y * 64;
  const int chunk = m0 >> 8;

  __shared__ __align__(16) __bf16 As[128 * LDA];
  __shared__ __align__(16) __bf16 Bs[ 64 * LDA];

  v8f acc[2][2] = {};
  const int l16 = lane & 15;
  const int sel = lane >> 4;

  for (int phase = 0; phase < 2; ++phase) {
    const __bf16* A = phase ? act_ex : act_sh;
    const float*  B = phase ? (Wexp_base + (int64_t)chunk * I_DIM * H_DIM) : Wsh;

    for (int k0 = 0; k0 < I_DIM; k0 += 32) {
      __syncthreads();
#pragma unroll
      for (int it = 0; it < 2; ++it) {
        const int row = (tid >> 2) + it * 64;
        const int c8  = (tid & 3) * 8;
        copy_b128_to_lds(A + (int64_t)(m0 + row) * I_DIM + k0 + c8,
                         &As[row * LDA + c8]);
      }
      {
        const int k2 = (tid >> 4) * 2;
        const int n  = (tid & 15) * 4;
        const int64_t g0 = (int64_t)(k0 + k2) * H_DIM + n0 + n;
        const float4 ba = *(const float4*)(B + g0);
        const float4 bb = *(const float4*)(B + g0 + H_DIM);
        *(uint32_t*)(&Bs[(n + 0) * LDA + k2]) = pack_bf16x2(ba.x, bb.x);
        *(uint32_t*)(&Bs[(n + 1) * LDA + k2]) = pack_bf16x2(ba.y, bb.y);
        *(uint32_t*)(&Bs[(n + 2) * LDA + k2]) = pack_bf16x2(ba.z, bb.z);
        *(uint32_t*)(&Bs[(n + 3) * LDA + k2]) = pack_bf16x2(ba.w, bb.w);
        if (k0 + 32 < I_DIM) {
          __builtin_prefetch(B + g0 + 32 * H_DIM, 0, 1);
        }
      }
      wait_async_lds();
      __syncthreads();

      v16bf afrag[2], bfrag[2];
#pragma unroll
      for (int mi = 0; mi < 2; ++mi) {
        const int base = (wm * 32 + mi * 16 + l16) * LDA + sel * 8;
        afrag[mi] = cat8(*(const v8bf*)(&As[base]), *(const v8bf*)(&As[base + 16]));
      }
#pragma unroll
      for (int ni = 0; ni < 2; ++ni) {
        const int base = (wn * 32 + ni * 16 + l16) * LDA + sel * 8;
        bfrag[ni] = cat8(*(const v8bf*)(&Bs[base]), *(const v8bf*)(&Bs[base + 16]));
      }
#pragma unroll
      for (int mi = 0; mi < 2; ++mi) {
#pragma unroll
        for (int ni = 0; ni < 2; ++ni) {
          acc[mi][ni] = __builtin_amdgcn_wmma_f32_16x16x32_bf16(
              false, afrag[mi], false, bfrag[ni], (short)0, acc[mi][ni], false, false);
        }
      }
    }
  }

#pragma unroll
  for (int mi = 0; mi < 2; ++mi) {
#pragma unroll
    for (int ni = 0; ni < 2; ++ni) {
#pragma unroll
      for (int r = 0; r < 8; ++r) {
        const int m = m0 + wm * 32 + mi * 16 + r + 8 * sel;
        const int n = n0 + wn * 32 + ni * 16 + l16;
        out[(int64_t)m * H_DIM + n] = acc[mi][ni][r];
      }
    }
  }
}

// ---------------------------------------------------------------------------
extern "C" void kernel_launch(void* const* d_in, const int* in_sizes, int n_in,
                              void* d_out, int out_size, void* d_ws, size_t ws_size,
                              hipStream_t stream) {
  (void)in_sizes; (void)n_in; (void)out_size; (void)ws_size;
  const float* x   = (const float*)d_in[0];  // (2,1024,1024)
  const float* rw  = (const float*)d_in[1];  // (1024,8)
  const float* gup = (const float*)d_in[2];  // (8,1024,8192)
  const float* dwn = (const float*)d_in[3];  // (8,4096,1024)
  const float* sg  = (const float*)d_in[4];  // (1024,4096)
  const float* su  = (const float*)d_in[5];  // (1024,4096)
  const float* sd  = (const float*)d_in[6];  // (4096,1024)
  float* out = (float*)d_out;                // [T*H final | E*T router_scores.T]

  char* ws = (char*)d_ws;
  float*  scores = (float*)ws;                                  //   8 KB
  __bf16* x16    = (__bf16*)(ws + 16384);                       //   4 MB
  __bf16* xs16   = x16 + (size_t)T_TOK * H_DIM;                 //   4 MB
  __bf16* act_sh = xs16 + (size_t)T_TOK * H_DIM;                //  16 MB
  __bf16* act_ex = act_sh + (size_t)T_TOK * I_DIM;              //  16 MB

  // 1) router: scores + router_scores.T (output tail)
  router_kernel<<<T_TOK / 8, 256, 0, stream>>>(
      x, rw, scores, out + (size_t)T_TOK * H_DIM);
  // 2) bf16 conversion of activations (plain and router-scaled)
  convert_kernel<<<(T_TOK * H_DIM / 4) / 256, 256, 0, stream>>>(
      x, scores, x16, xs16);
  // 3) shared MLP gate/up + SwiGLU
  gemm_gateup_kernel<<<dim3(T_TOK / 128, I_DIM / 64), 256, 0, stream>>>(
      x16, sg, su, (int64_t)I_DIM, (int64_t)0, act_sh);
  // 4) expert gate/up + SwiGLU (per-chunk expert weights; gate = cols [0,I), up = [I,2I))
  gemm_gateup_kernel<<<dim3(T_TOK / 128, I_DIM / 64), 256, 0, stream>>>(
      xs16, gup, gup + I_DIM, (int64_t)(2 * I_DIM), (int64_t)H_DIM * 2 * I_DIM, act_ex);
  // 5) down projections (shared + expert) summed -> final fp32 output
  gemm_down_kernel<<<dim3(T_TOK / 128, H_DIM / 64), 256, 0, stream>>>(
      act_sh, act_ex, sd, dwn, out);
}
